// BaseGIN_69990787056151
// MI455X (gfx1250) — matrix operand order repaired and checked by reading, not compiled, not run on hardware
//
#include <hip/hip_runtime.h>
#include <hip/hip_bf16.h>

typedef __attribute__((ext_vector_type(16))) __bf16 v16bf;
typedef __attribute__((ext_vector_type(2)))  __bf16 bf16x2;
typedef __attribute__((ext_vector_type(8)))  float  v8f;

static constexpr int HD = 64;          // hidden width (all layer outputs)
static constexpr float kBnEps = 1e-5f;

// Packed f32x2 -> bf16x2 conversion (hardware v_cvt_pk_bf16_f32 when available).
__device__ __forceinline__ unsigned pack2bf(float lo, float hi) {
#if __has_builtin(__builtin_amdgcn_cvt_pk_bf16_f32)
  bf16x2 p = __builtin_amdgcn_cvt_pk_bf16_f32(lo, hi);
  return __builtin_bit_cast(unsigned, p);
#else
  unsigned short lb = __builtin_bit_cast(unsigned short, (__bf16)lo);
  unsigned short hb = __builtin_bit_cast(unsigned short, (__bf16)hi);
  return (unsigned)lb | ((unsigned)hb << 16);
#endif
}

// Load a 16-element bf16 fragment from two 16-byte-aligned LDS runs of 8.
__device__ __forceinline__ v16bf load_frag16(const __bf16* p0, const __bf16* p1) {
  struct QQ { uint4 a, b; } q;
  q.a = *(const uint4*)p0;
  q.b = *(const uint4*)p1;
  return __builtin_bit_cast(v16bf, q);
}

// ---------------------------------------------------------------- zero fill
__global__ void zero_kernel(float* __restrict__ p, long long n) {
  long long t = (long long)blockIdx.x * blockDim.x + threadIdx.x;
  long long stride = (long long)gridDim.x * blockDim.x;
  for (; t < n; t += stride) p[t] = 0.0f;
}

// ------------------------------------------------- edge-parallel scatter-add
// agg[dst[e]][f] += ew[e] * x[src[e]][f]; one thread handles 4 features.
template<int F>
__global__ void scatter_kernel(const float* __restrict__ x,
                               const long long* __restrict__ src,
                               const long long* __restrict__ dst,
                               const float* __restrict__ ew,
                               float* __restrict__ agg, int E) {
  constexpr int VPE = F / 4;
  long long t = (long long)blockIdx.x * blockDim.x + threadIdx.x;
  if (t >= (long long)E * VPE) return;
  int e  = (int)(t / VPE);
  int f4 = (int)(t % VPE) * 4;
  long long s = src[e];
  long long d = dst[e];
  float w = ew[e];
  const float4 xv = *(const float4*)(x + s * F + f4);
  float* ap = agg + d * F + f4;
  atomicAdd(ap + 0, xv.x * w);
  atomicAdd(ap + 1, xv.y * w);
  atomicAdd(ap + 2, xv.z * w);
  atomicAdd(ap + 3, xv.w * w);
}

// ------------------------------------------- fused GIN MLP (two WMMA GEMMs)
// h    = (1+eps)*x + agg                       [N,F]
// mid  = relu(h @ Wa + ba)                     [N,64]  (kept in LDS, bf16)
// out  = mid @ Wb + bb                         [N,64]  (+ per-channel sum/sumsq)
template<int F>
__global__ __launch_bounds__(128)
void gin_mlp_kernel(const float* __restrict__ x,
                    const float* __restrict__ agg,
                    const float* __restrict__ epsp,
                    const float* __restrict__ Wa, const float* __restrict__ ba,
                    const float* __restrict__ Wb, const float* __restrict__ bb,
                    float* __restrict__ out,
                    float* __restrict__ gsum, float* __restrict__ gsumsq,
                    int N) {
  constexpr int MSTR = 72;                       // smid row stride (16B-aligned runs)
  __shared__ alignas(16) __bf16 sWaT[HD * F];    // transposed: [n][k]
  __shared__ alignas(16) __bf16 sWbT[HD * HD];   // transposed: [n][k]
  __shared__ alignas(16) __bf16 smid[64 * MSTR]; // mid activations, bf16
  __shared__ float ssum[HD], ssumsq[HD];

  const int tid  = threadIdx.x;
  const int lane = tid & 31;
  const int wave = tid >> 5;

  // Stage weights into LDS transposed, packed two-bf16 per b32 store.
  {
    unsigned* wa32 = (unsigned*)sWaT;
    for (int i = tid; i < HD * F / 2; i += 128) {
      int n  = i / (F / 2);
      int k2 = (i - n * (F / 2)) * 2;
      wa32[i] = pack2bf(Wa[(k2 + 0) * HD + n], Wa[(k2 + 1) * HD + n]);
    }
    unsigned* wb32 = (unsigned*)sWbT;
    for (int i = tid; i < HD * HD / 2; i += 128) {
      int n  = i / (HD / 2);
      int k2 = (i - n * (HD / 2)) * 2;
      wb32[i] = pack2bf(Wb[(k2 + 0) * HD + n], Wb[(k2 + 1) * HD + n]);
    }
  }
  if (tid < HD) { ssum[tid] = 0.0f; ssumsq[tid] = 0.0f; }
  __syncthreads();

  const float ep1   = 1.0f + epsp[0];
  const int m       = lane & 15;          // A: row within 16-row strip
  const int koff    = (lane >> 4) * 8;    // A: K sub-offset per lane half
  const int nloc    = lane & 15;          // B: column within 16
  const int ksel    = (lane >> 4) * 8;    // B: K sub-offset per lane half
  const int rowbase = blockIdx.x * 64 + wave * 16;
  const int row     = rowbase + m;
  const bool rowok  = row < N;

  v8f c1[4] = {};
  v8f c2[4] = {};

  // ---- stage 1: [16,F] x [F,64] ----
  const float* xr = x   + (long long)row * F;
  const float* ar = agg + (long long)row * F;
  #pragma unroll
  for (int kt = 0; kt < F / 32; ++kt) {
    struct U8 { unsigned w[8]; } u;
    if (rowok) {
      const int base = kt * 32 + koff;
      #pragma unroll
      for (int i = 0; i < 4; ++i) {
        int k0 = base + 2 * i;
        int k1 = base + 16 + 2 * i;
        u.w[i]     = pack2bf(ep1 * xr[k0] + ar[k0], ep1 * xr[k0 + 1] + ar[k0 + 1]);
        u.w[4 + i] = pack2bf(ep1 * xr[k1] + ar[k1], ep1 * xr[k1 + 1] + ar[k1 + 1]);
      }
    } else {
      #pragma unroll
      for (int i = 0; i < 8; ++i) u.w[i] = 0u;
    }
    v16bf a = __builtin_bit_cast(v16bf, u);
    #pragma unroll
    for (int nt = 0; nt < 4; ++nt) {
      const __bf16* bp = &sWaT[(nt * 16 + nloc) * F + kt * 32 + ksel];
      v16bf b = load_frag16(bp, bp + 16);
      c1[nt] = __builtin_amdgcn_wmma_f32_16x16x32_bf16(
          false, a, false, b, (short)0, c1[nt], false, false);
    }
  }

  // bias + relu -> LDS mid (bf16)
  #pragma unroll
  for (int nt = 0; nt < 4; ++nt) {
    #pragma unroll
    for (int r = 0; r < 8; ++r) {
      int M  = r + (lane >> 4) * 8;
      int nn = nt * 16 + nloc;
      float v = c1[nt][r] + ba[nn];
      v = fmaxf(v, 0.0f);
      smid[(wave * 16 + M) * MSTR + nn] = (__bf16)v;
    }
  }
  __syncthreads();

  // ---- stage 2: [16,64] x [64,64] ----
  #pragma unroll
  for (int kt = 0; kt < 2; ++kt) {
    const __bf16* ap0 = &smid[(wave * 16 + m) * MSTR + kt * 32 + koff];
    v16bf a = load_frag16(ap0, ap0 + 16);
    #pragma unroll
    for (int nt = 0; nt < 4; ++nt) {
      const __bf16* bp = &sWbT[(nt * 16 + nloc) * HD + kt * 32 + ksel];
      v16bf b = load_frag16(bp, bp + 16);
      c2[nt] = __builtin_amdgcn_wmma_f32_16x16x32_bf16(
          false, a, false, b, (short)0, c2[nt], false, false);
    }
  }

  // bias, store, per-channel stats (register-reduce rows, then LDS, then global)
  #pragma unroll
  for (int nt = 0; nt < 4; ++nt) {
    int nn = nt * 16 + nloc;
    float s = 0.0f, s2 = 0.0f;
    #pragma unroll
    for (int r = 0; r < 8; ++r) {
      int M    = r + (lane >> 4) * 8;
      int orow = rowbase + M;
      if (orow < N) {
        float v = c2[nt][r] + bb[nn];
        out[(long long)orow * HD + nn] = v;
        s  += v;
        s2 += v * v;
      }
    }
    atomicAdd(&ssum[nn], s);
    atomicAdd(&ssumsq[nn], s2);
  }
  __syncthreads();
  if (tid < HD) {
    atomicAdd(&gsum[tid], ssum[tid]);
    atomicAdd(&gsumsq[tid], ssumsq[tid]);
  }
}

// ------------------------------------------------ fold stats -> scale/shift
__global__ void bn_prep_kernel(const float* __restrict__ gsum,
                               const float* __restrict__ gsumsq,
                               const float* __restrict__ gamma,
                               const float* __restrict__ beta,
                               float* __restrict__ scale,
                               float* __restrict__ shift, int N) {
  int c = threadIdx.x;
  if (c >= HD) return;
  float invN = 1.0f / (float)N;
  float mean = gsum[c] * invN;
  float var  = gsumsq[c] * invN - mean * mean;
  float sc   = gamma[c] * rsqrtf(var + kBnEps);
  scale[c] = sc;
  shift[c] = beta[c] - mean * sc;
}

// ---------------------------------------- BN apply + ReLU (+ residual merge)
__global__ void bn_apply_kernel(const float* __restrict__ h2,
                                const float* __restrict__ scale,
                                const float* __restrict__ shift,
                                float* __restrict__ hio,
                                long long total, int residual) {
  long long t = (long long)blockIdx.x * blockDim.x + threadIdx.x;
  if (t >= total) return;
  int c = (int)(t & (HD - 1));
  float v = fmaxf(h2[t] * scale[c] + shift[c], 0.0f);
  hio[t] = residual ? (hio[t] + v) : v;
}

// ---------------------------------------------------------------- dispatcher
extern "C" void kernel_launch(void* const* d_in, const int* in_sizes, int n_in,
                              void* d_out, int out_size, void* d_ws, size_t ws_size,
                              hipStream_t stream) {
  const float*     x    = (const float*)d_in[0];
  const long long* ei   = (const long long*)d_in[1];
  const float*     ew   = (const float*)d_in[2];
  const float*     eps1 = (const float*)d_in[3];
  const float*     W1a  = (const float*)d_in[4];
  const float*     b1a  = (const float*)d_in[5];
  const float*     W1b  = (const float*)d_in[6];
  const float*     b1b  = (const float*)d_in[7];
  const float*     g1   = (const float*)d_in[8];
  const float*     be1  = (const float*)d_in[9];
  const float*     epss = (const float*)d_in[10];
  const float*     Wsa  = (const float*)d_in[11];
  const float*     bsa  = (const float*)d_in[12];
  const float*     Wsb  = (const float*)d_in[13];
  const float*     bsb  = (const float*)d_in[14];
  const float*     gsA  = (const float*)d_in[15];
  const float*     besA = (const float*)d_in[16];

  const int F   = 128;
  const int N   = in_sizes[0] / F;
  const int E   = in_sizes[2];
  const int Lm1 = in_sizes[10];   // number of hidden->hidden layers

  const long long* srcp = ei;
  const long long* dstp = ei + E;

  char*  ws     = (char*)d_ws;
  float* agg    = (float*)ws;                                      // N*128 f32
  float* gh2    = (float*)(ws + (size_t)N * 128 * sizeof(float));  // N*64 f32
  float* stats  = (float*)(ws + (size_t)N * 128 * sizeof(float)
                              + (size_t)N * 64 * sizeof(float));   // 256 f32
  float* gsum   = stats;
  float* gsumsq = stats + 64;
  float* scale  = stats + 128;
  float* shift  = stats + 192;
  float* hout   = (float*)d_out;   // running node features [N,64]

  const int gemm_blocks = (N + 63) / 64;
  const long long nh = (long long)N * HD;
  const int bn_blocks = (int)((nh + 255) / 256);

  // ---------------- layer 1 (F=128 -> 64) ----------------
  zero_kernel<<<2048, 256, 0, stream>>>(agg, (long long)N * 128);
  zero_kernel<<<1, 256, 0, stream>>>(stats, 256);
  {
    long long tot = (long long)E * (128 / 4);
    scatter_kernel<128><<<(int)((tot + 255) / 256), 256, 0, stream>>>(
        x, srcp, dstp, ew, agg, E);
  }
  gin_mlp_kernel<128><<<gemm_blocks, 128, 0, stream>>>(
      x, agg, eps1, W1a, b1a, W1b, b1b, gh2, gsum, gsumsq, N);
  bn_prep_kernel<<<1, 64, 0, stream>>>(gsum, gsumsq, g1, be1, scale, shift, N);
  bn_apply_kernel<<<bn_blocks, 256, 0, stream>>>(gh2, scale, shift, hout, nh, 0);

  // ---------------- layers 2..L (64 -> 64, residual) ----------------
  for (int l = 0; l < Lm1; ++l) {
    zero_kernel<<<2048, 256, 0, stream>>>(agg, (long long)N * 64);
    zero_kernel<<<1, 256, 0, stream>>>(stats, 256);
    {
      long long tot = (long long)E * (64 / 4);
      scatter_kernel<64><<<(int)((tot + 255) / 256), 256, 0, stream>>>(
          hout, srcp, dstp, ew, agg, E);
    }
    gin_mlp_kernel<64><<<gemm_blocks, 128, 0, stream>>>(
        hout, agg, epss + l,
        Wsa + (size_t)l * 64 * 64, bsa + (size_t)l * 64,
        Wsb + (size_t)l * 64 * 64, bsb + (size_t)l * 64,
        gh2, gsum, gsumsq, N);
    bn_prep_kernel<<<1, 64, 0, stream>>>(gsum, gsumsq,
                                         gsA + (size_t)l * 64,
                                         besA + (size_t)l * 64,
                                         scale, shift, N);
    bn_apply_kernel<<<bn_blocks, 256, 0, stream>>>(gh2, scale, shift, hout, nh, 1);
  }
}
